// PythiaAttention_65326452572218
// MI455X (gfx1250) — matrix operand order repaired and checked by reading, not compiled, not run on hardware
//
#include <hip/hip_runtime.h>
#include <hip/hip_bf16.h>

#define HIDDEN 1024
#define SEQ    2048
#define BATCH  4
#define NHEADS 16
#define HDIM   64
#define RDIM   16
#define NQKV   (3 * HIDDEN)

typedef __bf16 bf16_t;
typedef bf16_t v16bf __attribute__((ext_vector_type(16)));
typedef float  v8f   __attribute__((ext_vector_type(8)));
typedef int    v4i_  __attribute__((ext_vector_type(4)));

union FragBF {
  v16bf v;
  uint4 u[2];
};

// ---- async global->LDS copy path (CDNA5), with portable fallback ----------
#if defined(__has_builtin)
#if __has_builtin(__builtin_amdgcn_global_load_async_to_lds_b128) && \
    __has_builtin(__builtin_amdgcn_s_wait_asynccnt)
#define HAVE_ASYNC_LDS 1
#endif
#endif
#ifndef HAVE_ASYNC_LDS
#define HAVE_ASYNC_LDS 0
#endif

__device__ __forceinline__ void copy16B_to_lds(const unsigned short* g, unsigned short* l) {
#if HAVE_ASYNC_LDS
  // signature (from clang diagnostic): (v4i AS1*, v4i AS3*, i32 offset, i32 cpol)
  __builtin_amdgcn_global_load_async_to_lds_b128(
      (__attribute__((address_space(1))) v4i_*)g,
      (__attribute__((address_space(3))) v4i_*)l, 0, 0);
#else
  *(uint4*)l = *(const uint4*)g;
#endif
}

__device__ __forceinline__ void async_stage_wait() {
#if HAVE_ASYNC_LDS
  __builtin_amdgcn_s_wait_asynccnt(0);
#endif
}

__device__ __forceinline__ unsigned short f2bf(float x) {
  unsigned int u = __float_as_uint(x);
  u += 0x7FFFu + ((u >> 16) & 1u);          // round-to-nearest-even
  return (unsigned short)(u >> 16);
}

// A operand (16x32 bf16): lane m = L%16; elements = K[kb..kb+7], K[kb+16..kb+23], kb = 8*(L/16)
__device__ __forceinline__ v16bf lds_fragA(const unsigned short* rowptr, int kb) {
  FragBF f;
  f.u[0] = *(const uint4*)(rowptr + kb);
  f.u[1] = *(const uint4*)(rowptr + kb + 16);
  return f.v;
}

// B operand (32x16 bf16): lane n = L%16; elements = K[ks..ks+7], K[ks+16..ks+23], ks = 8*(L/16)
__device__ __forceinline__ v16bf lds_fragB(const unsigned short* rowptr, int ks) {
  FragBF f;
  f.u[0] = *(const uint4*)(rowptr + ks);
  f.u[1] = *(const uint4*)(rowptr + ks + 16);
  return f.v;
}

#define WMMA_BF16(A, B, C) \
  __builtin_amdgcn_wmma_f32_16x16x32_bf16(false, (A), false, (B), (short)0, (C), false, false)

// -------------------------------------------------------------------------
// Kernel 0: fp32 -> bf16 convert, optionally transposed ([k][n] -> [n][k]).
// Bandwidth-trivial one-shot pass so GEMM staging becomes raw b128 copies.
// -------------------------------------------------------------------------
__global__ __launch_bounds__(256)
void cvt_bf16_kernel(const float* __restrict__ src, unsigned short* __restrict__ dst,
                     int rows, int cols, int transpose) {
  size_t n = (size_t)rows * cols;
  for (size_t i = (size_t)blockIdx.x * 256 + threadIdx.x; i < n;
       i += (size_t)gridDim.x * 256) {
    unsigned short v = f2bf(src[i]);
    if (transpose) {
      int r = (int)(i / cols);
      int c = (int)(i % cols);
      dst[(size_t)c * rows + r] = v;
    } else {
      dst[i] = v;
    }
  }
}

// -------------------------------------------------------------------------
// Kernel 1: QKV = X @ Wqkv + b, fused partial RoPE.
// A = Xbf [8192x1024] bf16, B = WqkvT [3072][1024] bf16 (pre-transposed).
// Q,K scattered to [b,h,s,d]; V scattered to [b,h,d,s] (transposed) so the
// attention kernel's P.V B-operand stages as contiguous b128 copies.
// -------------------------------------------------------------------------
__global__ __launch_bounds__(256)
void qkv_rope_kernel(const unsigned short* __restrict__ Xbf,
                     const unsigned short* __restrict__ WT,
                     const float* __restrict__ bias,
                     unsigned short* __restrict__ Qo, unsigned short* __restrict__ Ko,
                     unsigned short* __restrict__ Vt) {
  __shared__ unsigned short sA[128 * 40];   // [row][k], 80B rows (16B aligned)
  __shared__ unsigned short sB[64 * 40];    // [n][k]

  const int tid  = threadIdx.x;
  const int lane = tid & 31;
  const int l16  = lane & 15;
  const int kb   = (lane >> 4) * 8;
  const int ks   = (lane >> 4) * 16;
  const int wave = tid >> 5;
  const int wm   = wave & 3;
  const int wn   = wave >> 2;
  const int m0   = blockIdx.y * 128;
  const int n0   = blockIdx.x * 64;

  v8f z = {0.f, 0.f, 0.f, 0.f, 0.f, 0.f, 0.f, 0.f};
  v8f acc[2][2] = {{z, z}, {z, z}};

  const int arow = tid >> 1;            // 0..127
  const int ac   = (tid & 1) * 16;      // 0 / 16
  const int brow = tid >> 2;            // 0..63
  const int bc   = (tid & 3) * 8;       // 0,8,16,24

  for (int k0 = 0; k0 < HIDDEN; k0 += 32) {
    copy16B_to_lds(Xbf + (size_t)(m0 + arow) * HIDDEN + k0 + ac,     &sA[arow * 40 + ac]);
    copy16B_to_lds(Xbf + (size_t)(m0 + arow) * HIDDEN + k0 + ac + 8, &sA[arow * 40 + ac + 8]);
    copy16B_to_lds(WT  + (size_t)(n0 + brow) * HIDDEN + k0 + bc,     &sB[brow * 40 + bc]);
    async_stage_wait();
    __syncthreads();

    v16bf a0 = lds_fragA(&sA[(wm * 32 +  0 + l16) * 40], kb);
    v16bf a1 = lds_fragA(&sA[(wm * 32 + 16 + l16) * 40], kb);
    v16bf b0 = lds_fragB(&sB[(wn * 32 +  0 + l16) * 40], kb);
    v16bf b1 = lds_fragB(&sB[(wn * 32 + 16 + l16) * 40], kb);
    acc[0][0] = WMMA_BF16(a0, b0, acc[0][0]);
    acc[0][1] = WMMA_BF16(a0, b1, acc[0][1]);
    acc[1][0] = WMMA_BF16(a1, b0, acc[1][0]);
    acc[1][1] = WMMA_BF16(a1, b1, acc[1][1]);
    __syncthreads();
  }

  // epilogue: bias + partial RoPE + scatter
  const float LN1E4_OVER_8 = 1.1512925465f;   // ln(10000)/8
#pragma unroll
  for (int sm = 0; sm < 2; ++sm) {
#pragma unroll
    for (int sn = 0; sn < 2; ++sn) {
      int colbase = n0 + wn * 32 + sn * 16;
#pragma unroll
      for (int v = 0; v < 8; ++v) {
        int row = m0 + wm * 32 + sm * 16 + v + 8 * (lane >> 4);
        int col = colbase + l16;
        float val = acc[sm][sn][v] + bias[col];
        float partner = __shfl_xor(val, 8, 32);   // column col^8, same row
        int d = col & (HDIM - 1);
        int three = col >> 10;                    // 0=q 1=k 2=v
        if (three < 2 && d < RDIM) {
          int s = row & (SEQ - 1);
          int j = d & 7;
          float freq = __expf(-LN1E4_OVER_8 * (float)j);
          float ang = (float)s * freq;
          float sv, cv;
          __sincosf(ang, &sv, &cv);
          float rot = (d & 8) ? partner : -partner;   // rotate_half
          val = val * cv + rot * sv;
        }
        int b = row >> 11;
        int s = row & (SEQ - 1);
        int head = (col & (HIDDEN - 1)) >> 6;
        int bh = b * NHEADS + head;
        if (three == 2) {
          Vt[((size_t)(bh * HDIM + d)) * SEQ + s] = f2bf(val);     // [b,h,d,s]
        } else {
          unsigned short* dst = three ? Ko : Qo;
          dst[((size_t)(bh * SEQ + s)) * HDIM + d] = f2bf(val);    // [b,h,s,d]
        }
      }
    }
  }
}

// -------------------------------------------------------------------------
// Kernel 2: causal flash attention, one block per (64-row q tile, b*h)
// -------------------------------------------------------------------------
__global__ __launch_bounds__(256)
void attn_kernel(const unsigned short* __restrict__ Qg, const unsigned short* __restrict__ Kg,
                 const unsigned short* __restrict__ Vt, unsigned short* __restrict__ AO) {
  __shared__ unsigned short sQ[64 * 72];   // [row][d]
  __shared__ unsigned short sK[64 * 72];   // [key][d]  (B operand for Q.K^T)
  __shared__ unsigned short sV[64 * 72];   // [d][key]  (B operand for P.V)
  __shared__ unsigned short sP[64 * 72];   // [row][key] bf16 probabilities (A operand)
  __shared__ float          sS[64 * 66];   // raw scores
  __shared__ float          sAux[64];      // alpha, then 1/l

  const int tid  = threadIdx.x;
  const int lane = tid & 31;
  const int l16  = lane & 15;
  const int kb   = (lane >> 4) * 8;
  const int ks   = (lane >> 4) * 16;
  const int wave = tid >> 5;
  const int mt   = wave & 3;
  const int nt0  = (wave >> 2) * 2;
  const int qb   = blockIdx.x;
  const int bh   = blockIdx.y;
  const int q0   = qb * 64;
  const size_t base  = (size_t)bh * SEQ * HDIM;     // Q/K base
  const size_t vbase = (size_t)bh * HDIM * SEQ;     // Vt base

  const int r  = tid >> 2;            // 0..63
  const int d0 = (tid & 3) * 16;      // 0,16,32,48

  // stage Q tile once
  copy16B_to_lds(Qg + base + (size_t)(q0 + r) * HDIM + d0,     &sQ[r * 72 + d0]);
  copy16B_to_lds(Qg + base + (size_t)(q0 + r) * HDIM + d0 + 8, &sQ[r * 72 + d0 + 8]);

  v8f z = {0.f, 0.f, 0.f, 0.f, 0.f, 0.f, 0.f, 0.f};
  v8f o0 = z, o1 = z;
  float m_i = -__builtin_inff();
  float l_i = 0.f;

  for (int kt = 0; kt <= qb; ++kt) {
    // stage K [key][d]; V tile direct from pre-transposed Vt -> sV[d][key]
    copy16B_to_lds(Kg + base + (size_t)(kt * 64 + r) * HDIM + d0,       &sK[r * 72 + d0]);
    copy16B_to_lds(Kg + base + (size_t)(kt * 64 + r) * HDIM + d0 + 8,   &sK[r * 72 + d0 + 8]);
    copy16B_to_lds(Vt + vbase + (size_t)r * SEQ + kt * 64 + d0,         &sV[r * 72 + d0]);
    copy16B_to_lds(Vt + vbase + (size_t)r * SEQ + kt * 64 + d0 + 8,     &sV[r * 72 + d0 + 8]);
    if (kt < qb) {  // prefetch next K tile (global_prefetch_b8)
      __builtin_prefetch(Kg + base + (size_t)((kt + 1) * 64 + r) * HDIM + d0, 0, 0);
    }
    async_stage_wait();
    __syncthreads();

    // S = Q.K^T  (M=64 q rows, N=64 keys, K=d=64)
    v8f s0 = z, s1 = z;
#pragma unroll
    for (int kd = 0; kd < 64; kd += 32) {
      v16bf a  = lds_fragA(&sQ[(mt * 16 + l16) * 72 + kd], kb);
      v16bf b0 = lds_fragB(&sK[((nt0    ) * 16 + l16) * 72 + kd], kb);
      v16bf b1 = lds_fragB(&sK[((nt0 + 1) * 16 + l16) * 72 + kd], kb);
      s0 = WMMA_BF16(a, b0, s0);
      s1 = WMMA_BF16(a, b1, s1);
    }
    // scale + causal mask -> sS
#pragma unroll
    for (int j = 0; j < 2; ++j) {
      v8f sc = j ? s1 : s0;
      int klocal = (nt0 + j) * 16 + l16;
      int key = kt * 64 + klocal;
#pragma unroll
      for (int v = 0; v < 8; ++v) {
        int rr = mt * 16 + v + 8 * (lane >> 4);
        float val = sc[v] * 0.125f;               // 1/sqrt(64)
        if (key > q0 + rr) val = -__builtin_inff();
        sS[rr * 66 + klocal] = val;
      }
    }
    __syncthreads();

    // online softmax: thread rr owns row rr; running state in registers
    if (tid < 64) {
      const float* rowp = &sS[tid * 66];
      float mx = m_i;
      for (int k = 0; k < 64; ++k) mx = fmaxf(mx, rowp[k]);
      float alpha = __expf(m_i - mx);
      float sum = 0.f;
      unsigned short* pp = &sP[tid * 72];
      for (int k = 0; k < 64; ++k) {
        float p = __expf(rowp[k] - mx);
        pp[k] = f2bf(p);
        sum += p;
      }
      l_i = l_i * alpha + sum;
      m_i = mx;
      sAux[tid] = alpha;
    }
    __syncthreads();

    // O = O*alpha + P.V   (M=q rows, N=d, K=keys=64)
#pragma unroll
    for (int v = 0; v < 8; ++v) {
      int rr = mt * 16 + v + 8 * (lane >> 4);
      float al = sAux[rr];
      o0[v] *= al;
      o1[v] *= al;
    }
#pragma unroll
    for (int kd = 0; kd < 64; kd += 32) {
      v16bf a  = lds_fragA(&sP[(mt * 16 + l16) * 72 + kd], kb);
      v16bf b0 = lds_fragB(&sV[((nt0    ) * 16 + l16) * 72 + kd], kb);
      v16bf b1 = lds_fragB(&sV[((nt0 + 1) * 16 + l16) * 72 + kd], kb);
      o0 = WMMA_BF16(a, b0, o0);
      o1 = WMMA_BF16(a, b1, o1);
    }
    __syncthreads();
  }

  if (tid < 64) sAux[tid] = 1.0f / l_i;
  __syncthreads();

  const int b = bh >> 4, h = bh & (NHEADS - 1);
#pragma unroll
  for (int j = 0; j < 2; ++j) {
    v8f oc = j ? o1 : o0;
    int d = (nt0 + j) * 16 + l16;
#pragma unroll
    for (int v = 0; v < 8; ++v) {
      int rr = mt * 16 + v + 8 * (lane >> 4);
      float val = oc[v] * sAux[rr];
      AO[((size_t)(b * SEQ + q0 + rr)) * HIDDEN + h * HDIM + d] = f2bf(val);
    }
  }
}

// -------------------------------------------------------------------------
// Kernel 3: Out = AO @ Wd + bd, fp32 output. B = WdT [n][k] bf16.
// -------------------------------------------------------------------------
__global__ __launch_bounds__(256)
void dense_kernel(const unsigned short* __restrict__ A, const unsigned short* __restrict__ WT,
                  const float* __restrict__ bias, float* __restrict__ Out) {
  __shared__ unsigned short sA[128 * 40];
  __shared__ unsigned short sB[64 * 40];

  const int tid  = threadIdx.x;
  const int lane = tid & 31;
  const int l16  = lane & 15;
  const int kb   = (lane >> 4) * 8;
  const int ks   = (lane >> 4) * 16;
  const int wave = tid >> 5;
  const int wm   = wave & 3;
  const int wn   = wave >> 2;
  const int m0   = blockIdx.y * 128;
  const int n0   = blockIdx.x * 64;

  v8f z = {0.f, 0.f, 0.f, 0.f, 0.f, 0.f, 0.f, 0.f};
  v8f acc[2][2] = {{z, z}, {z, z}};

  const int arow = tid >> 1;
  const int ac   = (tid & 1) * 16;
  const int brow = tid >> 2;
  const int bc   = (tid & 3) * 8;

  for (int k0 = 0; k0 < HIDDEN; k0 += 32) {
    copy16B_to_lds(A  + (size_t)(m0 + arow) * HIDDEN + k0 + ac,     &sA[arow * 40 + ac]);
    copy16B_to_lds(A  + (size_t)(m0 + arow) * HIDDEN + k0 + ac + 8, &sA[arow * 40 + ac + 8]);
    copy16B_to_lds(WT + (size_t)(n0 + brow) * HIDDEN + k0 + bc,     &sB[brow * 40 + bc]);
    async_stage_wait();
    __syncthreads();

    v16bf a0 = lds_fragA(&sA[(wm * 32 +  0 + l16) * 40], kb);
    v16bf a1 = lds_fragA(&sA[(wm * 32 + 16 + l16) * 40], kb);
    v16bf b0 = lds_fragB(&sB[(wn * 32 +  0 + l16) * 40], kb);
    v16bf b1 = lds_fragB(&sB[(wn * 32 + 16 + l16) * 40], kb);
    acc[0][0] = WMMA_BF16(a0, b0, acc[0][0]);
    acc[0][1] = WMMA_BF16(a0, b1, acc[0][1]);
    acc[1][0] = WMMA_BF16(a1, b0, acc[1][0]);
    acc[1][1] = WMMA_BF16(a1, b1, acc[1][1]);
    __syncthreads();
  }

#pragma unroll
  for (int sm = 0; sm < 2; ++sm) {
#pragma unroll
    for (int sn = 0; sn < 2; ++sn) {
      int colbase = n0 + wn * 32 + sn * 16;
#pragma unroll
      for (int v = 0; v < 8; ++v) {
        int row = m0 + wm * 32 + sm * 16 + v + 8 * (lane >> 4);
        int col = colbase + l16;
        Out[(size_t)row * HIDDEN + col] = acc[sm][sn][v] + bias[col];
      }
    }
  }
}

extern "C" void kernel_launch(void* const* d_in, const int* in_sizes, int n_in,
                              void* d_out, int out_size, void* d_ws, size_t ws_size,
                              hipStream_t stream) {
  (void)in_sizes; (void)n_in; (void)out_size; (void)ws_size;
  const float* X    = (const float*)d_in[0];
  const float* Wqkv = (const float*)d_in[1];
  const float* bqkv = (const float*)d_in[2];
  const float* Wd   = (const float*)d_in[3];
  const float* bd   = (const float*)d_in[4];
  float* out = (float*)d_out;

  const size_t NTOK  = (size_t)BATCH * SEQ;            // 8192
  const size_t NELEM = NTOK * HIDDEN;                  // 8M (per [B*S,H] tensor)
  unsigned short* Xbf   = (unsigned short*)d_ws;       // [8192][1024]
  unsigned short* WqkvT = Xbf + NELEM;                 // [3072][1024]
  unsigned short* WdT   = WqkvT + (size_t)NQKV * HIDDEN;   // [1024][1024]
  unsigned short* Q     = WdT + (size_t)HIDDEN * HIDDEN;   // [b,h,s,d]
  unsigned short* K     = Q + NELEM;                   // [b,h,s,d]
  unsigned short* Vt    = K + NELEM;                   // [b,h,d,s]
  unsigned short* AO    = Vt + NELEM;                  // [B*S, HIDDEN]

  // 0) one-shot fp32->bf16 conversion (+ weight transposes)
  cvt_bf16_kernel<<<2048, 256, 0, stream>>>(X, Xbf, (int)NTOK, HIDDEN, 0);
  cvt_bf16_kernel<<<2048, 256, 0, stream>>>(Wqkv, WqkvT, HIDDEN, NQKV, 1);
  cvt_bf16_kernel<<<1024, 256, 0, stream>>>(Wd, WdT, HIDDEN, HIDDEN, 1);
  // 1) QKV projection + bias + partial RoPE
  qkv_rope_kernel<<<dim3(NQKV / 64, (BATCH * SEQ) / 128), 256, 0, stream>>>(
      Xbf, WqkvT, bqkv, Q, K, Vt);
  // 2) causal flash attention
  attn_kernel<<<dim3(SEQ / 64, BATCH * NHEADS), 256, 0, stream>>>(Q, K, Vt, AO);
  // 3) output projection
  dense_kernel<<<dim3(HIDDEN / 64, (BATCH * SEQ) / 128), 256, 0, stream>>>(
      AO, WdT, bd, out);
}
